// GRUCell_27513560498404
// MI455X (gfx1250) — matrix operand order repaired and checked by reading, not compiled, not run on hardware
//
#include <hip/hip_runtime.h>
#include <hip/hip_bf16.h>

typedef __bf16 bf16_t;
typedef __attribute__((ext_vector_type(16))) __bf16 v16bf;
typedef __attribute__((ext_vector_type(8)))  __bf16 v8bf;
typedef __attribute__((ext_vector_type(8)))  float   v8f;

#define T_STEPS 512
#define BATCH   64
#define HID     1024
#define ROWS    (T_STEPS * BATCH)   // 32768
#define G3      (3 * HID)           // 3072
#define G2      (2 * HID)           // 2048
#define LN_EPS  1e-6f

// ---------------------------------------------------------------------------
// WMMA fragment loaders (CDNA5 wave32 layouts, CDNA5 ISA §7.12.2)
// A: 16x32 bf16.  lane<16: K in {0..7}∪{16..23}; lane>=16: {8..15}∪{24..31}
// B: 32x16 bf16 (B[k][n] = W[n][k]).  lane col=lane&15, same K split as A.
// ---------------------------------------------------------------------------
__device__ __forceinline__ v16bf load_a_frag(const bf16_t* tile, int ld) {
  const int lane = threadIdx.x & 31;
  const int row  = lane & 15;
  const int koff = (lane >> 4) << 3;   // 0 or 8
  const bf16_t* p = tile + (size_t)row * ld + koff;
  union { v16bf v; v8bf h[2]; } u;
  u.h[0] = *(const v8bf*)(p);          // K = koff .. koff+7
  u.h[1] = *(const v8bf*)(p + 16);     // K = koff+16 .. koff+23
  return u.v;
}

__device__ __forceinline__ v16bf load_b_frag(const bf16_t* wtile, int ld) {
  const int lane = threadIdx.x & 31;
  const int col  = lane & 15;
  const int kb   = (lane >> 4) << 3;   // 0 or 8
  const bf16_t* p = wtile + (size_t)col * ld + kb;
  union { v16bf v; v8bf h[2]; } u;
  u.h[0] = *(const v8bf*)(p);          // K = kb .. kb+7
  u.h[1] = *(const v8bf*)(p + 16);     // K = kb+16 .. kb+23
  return u.v;
}

__device__ __forceinline__ v8f wmma_bf16(v16bf a, v16bf b, v8f c) {
  return __builtin_amdgcn_wmma_f32_16x16x32_bf16(false, a, false, b,
                                                 (short)0, c, false, false);
}

__device__ __forceinline__ v8f vzero8() {
  v8f z;
#pragma unroll
  for (int i = 0; i < 8; ++i) z[i] = 0.0f;
  return z;
}

// ---------------------------------------------------------------------------
// fp32 -> bf16 conversion (vectorized x4)
// ---------------------------------------------------------------------------
__global__ void cvt_f32_bf16_kernel(const float* __restrict__ src,
                                    bf16_t* __restrict__ dst, int n4) {
  int i = blockIdx.x * blockDim.x + threadIdx.x;
  if (i >= n4) return;
  const float4 v = *(const float4*)(src + (size_t)i * 4);
  union { bf16_t b[4]; unsigned long long u; } o;
  o.b[0] = (bf16_t)v.x; o.b[1] = (bf16_t)v.y;
  o.b[2] = (bf16_t)v.z; o.b[3] = (bf16_t)v.w;
  *(unsigned long long*)(dst + (size_t)i * 4) = o.u;
}

__global__ void init_h_kernel(float* __restrict__ hf, bf16_t* __restrict__ hb, int n) {
  int i = blockIdx.x * blockDim.x + threadIdx.x;
  if (i < n) { hf[i] = 0.0f; hb[i] = (bf16_t)0.0f; }
}

// ---------------------------------------------------------------------------
// Input projection: gi = x_bf16 (32768x1024) @ w_ih^T (1024x3072), raw fp32.
// One wave = 16 rows x 64 cols (4 n-tiles).  The k-loop is unrolled x2 with
// ping-pong fragment buffers: while WMMAs consume set A, loads fill set B
// (and vice versa) -> no register rotation copies, no WAR hazard nops, and
// the k+32 loads overlap the k WMMAs (x and w_ih are both L2-resident).
// ---------------------------------------------------------------------------
#define GI_NT 4
__global__ void __launch_bounds__(256)
gemm_gi_kernel(const bf16_t* __restrict__ X, const bf16_t* __restrict__ W,
               float* __restrict__ gi) {
  const int wid = (blockIdx.x * blockDim.x + threadIdx.x) >> 5;
  const int NCH = G3 / (16 * GI_NT);               // 48
  const int mt  = wid / NCH;
  const int nc  = wid % NCH;
  const int m0  = mt * 16;
  const int n0  = nc * (16 * GI_NT);

  v8f acc[GI_NT];
#pragma unroll
  for (int t = 0; t < GI_NT; ++t) acc[t] = vzero8();

  const bf16_t* a_base = X + (size_t)m0 * HID;
  const bf16_t* w_base = W + (size_t)n0 * HID;

  v16bf aA = load_a_frag(a_base, HID);
  v16bf bA[GI_NT];
#pragma unroll
  for (int t = 0; t < GI_NT; ++t)
    bA[t] = load_b_frag(w_base + (size_t)t * 16 * HID, HID);

  for (int k0 = 0; k0 < HID; k0 += 64) {
    // fill set B for k0+32 while consuming set A at k0
    const v16bf aB = load_a_frag(a_base + k0 + 32, HID);
    v16bf bB[GI_NT];
#pragma unroll
    for (int t = 0; t < GI_NT; ++t)
      bB[t] = load_b_frag(w_base + (size_t)t * 16 * HID + k0 + 32, HID);
#pragma unroll
    for (int t = 0; t < GI_NT; ++t)
      acc[t] = wmma_bf16(aA, bA[t], acc[t]);
    // refill set A for k0+64 while consuming set B at k0+32
    if (k0 + 64 < HID) {
      aA = load_a_frag(a_base + k0 + 64, HID);
#pragma unroll
      for (int t = 0; t < GI_NT; ++t)
        bA[t] = load_b_frag(w_base + (size_t)t * 16 * HID + k0 + 64, HID);
    }
#pragma unroll
    for (int t = 0; t < GI_NT; ++t)
      acc[t] = wmma_bf16(aB, bB[t], acc[t]);
  }

  const int lane  = threadIdx.x & 31;
  const int col_l = lane & 15;
  const int rsel  = (lane >> 4) * 8;
#pragma unroll
  for (int t = 0; t < GI_NT; ++t) {
    const int col = n0 + t * 16 + col_l;
#pragma unroll
    for (int i = 0; i < 8; ++i)
      gi[(size_t)(m0 + rsel + i) * G3 + col] = acc[t][i];
  }
}

// ---------------------------------------------------------------------------
// Row-wise LN of gi (in place): cols [0,2048) with ln_ih, [2048,3072) ln_ih3.
// Bias b_ih added before the mean (gi3 gets the reference's scalar b_ih[2H]).
// One 256-thread block per row; deterministic tree reductions.
// ---------------------------------------------------------------------------
__global__ void __launch_bounds__(256)
ln_gi_kernel(float* __restrict__ gi, const float* __restrict__ b_ih,
             const float* __restrict__ g2, const float* __restrict__ bb2,
             const float* __restrict__ g3, const float* __restrict__ bb3) {
  const int row = blockIdx.x;
  const int tid = threadIdx.x;
  float* p = gi + (size_t)row * G3;
  __shared__ float sA[256], sA2[256], sB[256], sB2[256];
  __shared__ float st[4];

  const float b3s = b_ih[G2];            // scalar bias (reference quirk)
  float va[8], vb[4];
  float sa = 0.f, sa2 = 0.f, sb = 0.f, sb2 = 0.f;
#pragma unroll
  for (int j = 0; j < 8; ++j) {
    const int c = tid + j * 256;
    const float v = p[c] + b_ih[c];
    va[j] = v; sa += v; sa2 += v * v;
  }
#pragma unroll
  for (int j = 0; j < 4; ++j) {
    const int c = tid + j * 256;
    const float v = p[G2 + c] + b3s;
    vb[j] = v; sb += v; sb2 += v * v;
  }
  sA[tid] = sa; sA2[tid] = sa2; sB[tid] = sb; sB2[tid] = sb2;
  __syncthreads();
  for (int s = 128; s > 0; s >>= 1) {
    if (tid < s) {
      sA[tid] += sA[tid + s]; sA2[tid] += sA2[tid + s];
      sB[tid] += sB[tid + s]; sB2[tid] += sB2[tid + s];
    }
    __syncthreads();
  }
  if (tid == 0) {
    const float mA = sA[0] / (float)G2;
    const float vA = fmaxf((sA2[0] - (float)G2 * mA * mA) / (float)(G2 - 1), 0.f);
    st[0] = mA; st[1] = 1.0f / (sqrtf(vA) + LN_EPS);
    const float mB = sB[0] / (float)HID;
    const float vB = fmaxf((sB2[0] - (float)HID * mB * mB) / (float)(HID - 1), 0.f);
    st[2] = mB; st[3] = 1.0f / (sqrtf(vB) + LN_EPS);
  }
  __syncthreads();
  const float mA = st[0], iA = st[1], mB = st[2], iB = st[3];
#pragma unroll
  for (int j = 0; j < 8; ++j) {
    const int c = tid + j * 256;
    p[c] = g2[c] * (va[j] - mA) * iA + bb2[c];
  }
#pragma unroll
  for (int j = 0; j < 4; ++j) {
    const int c = tid + j * 256;
    p[G2 + c] = g3[c] * (vb[j] - mB) * iB + bb3[c];
  }
}

// ---------------------------------------------------------------------------
// Scan step part 1: gh2 = LN(h @ w_hh[:2H]^T + b_hh2); zr = sigmoid(gi2+gh2);
// writes z (fp32) and rh = r*h (bf16).  Grid = 4 blocks (16-row tiles of B),
// block = 512 threads = 16 waves, wave owns 128 cols (8 WMMA n-tiles).
// Loads for each k-step are grouped ahead of the 8 WMMAs, and the next
// k-block is warmed into WGP$ with global_prefetch (zero VGPR cost).
// ---------------------------------------------------------------------------
__global__ void __launch_bounds__(512)
step_zr_kernel(const bf16_t* __restrict__ hb, const bf16_t* __restrict__ whh,
               const float* __restrict__ gi, const float* __restrict__ b_hh,
               const float* __restrict__ lng, const float* __restrict__ lnb,
               const float* __restrict__ hf, float* __restrict__ zbuf,
               bf16_t* __restrict__ rhb, int t) {
  const int m0   = blockIdx.x * 16;
  const int wave = threadIdx.x >> 5;
  const int lane = threadIdx.x & 31;
  const int n0   = wave * 128;
  const int col_l = lane & 15;
  const int half  = lane >> 4;

  v8f acc[8];
#pragma unroll
  for (int nt = 0; nt < 8; ++nt) acc[nt] = vzero8();

  const bf16_t* a_base = hb + (size_t)m0 * HID;
  for (int k0 = 0; k0 < HID; k0 += 32) {
    if (k0 + 32 < HID) {  // prefetch next k-block into WGP$ (non-blocking)
      __builtin_prefetch(a_base + (size_t)col_l * HID + k0 + 32, 0, 1);
#pragma unroll
      for (int nt = 0; nt < 8; ++nt)
        __builtin_prefetch(whh + (size_t)(n0 + nt * 16 + col_l) * HID + k0 + 32, 0, 1);
    }
    const v16bf a = load_a_frag(a_base + k0, HID);
    v16bf b[8];
#pragma unroll
    for (int nt = 0; nt < 8; ++nt)
      b[nt] = load_b_frag(whh + (size_t)(n0 + nt * 16) * HID + k0, HID);
#pragma unroll
    for (int nt = 0; nt < 8; ++nt)
      acc[nt] = wmma_bf16(a, b[nt], acc[nt]);
  }
  // bias, per-lane row partials (sum / sumsq over this wave's 8 tiles)
  float s[8], s2[8];
#pragma unroll
  for (int i = 0; i < 8; ++i) { s[i] = 0.f; s2[i] = 0.f; }
#pragma unroll
  for (int nt = 0; nt < 8; ++nt) {
    const float bias = b_hh[n0 + nt * 16 + col_l];
#pragma unroll
    for (int i = 0; i < 8; ++i) {
      const float v = acc[nt][i] + bias;
      acc[nt][i] = v; s[i] += v; s2[i] += v * v;
    }
  }
  // reduce across the 16 column-lanes of each half-wave
#pragma unroll
  for (int off = 1; off < 16; off <<= 1) {
#pragma unroll
    for (int i = 0; i < 8; ++i) {
      s[i]  += __shfl_xor(s[i],  off, 32);
      s2[i] += __shfl_xor(s2[i], off, 32);
    }
  }
  __shared__ float red[16][16][2];
  __shared__ float stat[16][2];
  __shared__ float mrow[16], irow[16];
  if (col_l == 0) {              // lanes 0 and 16
#pragma unroll
    for (int i = 0; i < 8; ++i) {
      red[wave][half * 8 + i][0] = s[i];
      red[wave][half * 8 + i][1] = s2[i];
    }
  }
  __syncthreads();
  if (threadIdx.x < 32) {
    const int r = threadIdx.x & 15, c = threadIdx.x >> 4;
    float a = 0.f;
    for (int w = 0; w < 16; ++w) a += red[w][r][c];
    stat[r][c] = a;
  }
  __syncthreads();
  if (threadIdx.x < 16) {
    const int r = threadIdx.x;
    const float m = stat[r][0] / (float)G2;
    const float var = fmaxf((stat[r][1] - (float)G2 * m * m) / (float)(G2 - 1), 0.f);
    mrow[r] = m; irow[r] = 1.0f / (sqrtf(var) + LN_EPS);
  }
  __syncthreads();
  // epilogue: LN -> sigmoid(gi2+gh2) -> z / r*h
#pragma unroll
  for (int nt = 0; nt < 8; ++nt) {
    const int col = n0 + nt * 16 + col_l;
    const float gg = lng[col], gb = lnb[col];
#pragma unroll
    for (int i = 0; i < 8; ++i) {
      const int rl  = half * 8 + i;
      const int row = m0 + rl;
      const float vn = gg * (acc[nt][i] - mrow[rl]) * irow[rl] + gb;
      const float i2 = gi[((size_t)t * BATCH + row) * G3 + col];
      const float zr = 1.0f / (1.0f + expf(-(i2 + vn)));
      if (col < HID) {
        zbuf[row * HID + col] = zr;
      } else {
        const int j = col - HID;
        rhb[row * HID + j] = (bf16_t)(zr * hf[row * HID + j]);
      }
    }
  }
}

// ---------------------------------------------------------------------------
// Scan step part 2: gh3 = LN(rh @ w_hh[2H:]^T + b_hh3);
// h_new = (1-z)h + z*tanh(gi3+gh3).  Wave owns 64 cols (4 n-tiles),
// k-loop unrolled x2 with ping-pong fragment buffers (no rotation copies).
// ---------------------------------------------------------------------------
__global__ void __launch_bounds__(512)
step_h_kernel(const bf16_t* __restrict__ rhb, const bf16_t* __restrict__ whh3,
              const float* __restrict__ gi, const float* __restrict__ b_hh3,
              const float* __restrict__ lng, const float* __restrict__ lnb,
              const float* __restrict__ zbuf, float* __restrict__ hf,
              bf16_t* __restrict__ hb, float* __restrict__ out, int t) {
  const int m0   = blockIdx.x * 16;
  const int wave = threadIdx.x >> 5;
  const int lane = threadIdx.x & 31;
  const int n0   = wave * 64;
  const int col_l = lane & 15;
  const int half  = lane >> 4;

  v8f acc[4];
#pragma unroll
  for (int nt = 0; nt < 4; ++nt) acc[nt] = vzero8();

  const bf16_t* a_base = rhb + (size_t)m0 * HID;
  const bf16_t* w_base = whh3 + (size_t)n0 * HID;

  v16bf aA = load_a_frag(a_base, HID);
  v16bf bA[4];
#pragma unroll
  for (int nt = 0; nt < 4; ++nt)
    bA[nt] = load_b_frag(w_base + (size_t)nt * 16 * HID, HID);

  for (int k0 = 0; k0 < HID; k0 += 64) {
    const v16bf aB = load_a_frag(a_base + k0 + 32, HID);
    v16bf bB[4];
#pragma unroll
    for (int nt = 0; nt < 4; ++nt)
      bB[nt] = load_b_frag(w_base + (size_t)nt * 16 * HID + k0 + 32, HID);
#pragma unroll
    for (int nt = 0; nt < 4; ++nt)
      acc[nt] = wmma_bf16(aA, bA[nt], acc[nt]);
    if (k0 + 64 < HID) {
      aA = load_a_frag(a_base + k0 + 64, HID);
#pragma unroll
      for (int nt = 0; nt < 4; ++nt)
        bA[nt] = load_b_frag(w_base + (size_t)nt * 16 * HID + k0 + 64, HID);
    }
#pragma unroll
    for (int nt = 0; nt < 4; ++nt)
      acc[nt] = wmma_bf16(aB, bB[nt], acc[nt]);
  }

  float s[8], s2[8];
#pragma unroll
  for (int i = 0; i < 8; ++i) { s[i] = 0.f; s2[i] = 0.f; }
#pragma unroll
  for (int nt = 0; nt < 4; ++nt) {
    const float bias = b_hh3[n0 + nt * 16 + col_l];
#pragma unroll
    for (int i = 0; i < 8; ++i) {
      const float v = acc[nt][i] + bias;
      acc[nt][i] = v; s[i] += v; s2[i] += v * v;
    }
  }
#pragma unroll
  for (int off = 1; off < 16; off <<= 1) {
#pragma unroll
    for (int i = 0; i < 8; ++i) {
      s[i]  += __shfl_xor(s[i],  off, 32);
      s2[i] += __shfl_xor(s2[i], off, 32);
    }
  }
  __shared__ float red[16][16][2];
  __shared__ float stat[16][2];
  __shared__ float mrow[16], irow[16];
  if (col_l == 0) {
#pragma unroll
    for (int i = 0; i < 8; ++i) {
      red[wave][half * 8 + i][0] = s[i];
      red[wave][half * 8 + i][1] = s2[i];
    }
  }
  __syncthreads();
  if (threadIdx.x < 32) {
    const int r = threadIdx.x & 15, c = threadIdx.x >> 4;
    float a = 0.f;
    for (int w = 0; w < 16; ++w) a += red[w][r][c];
    stat[r][c] = a;
  }
  __syncthreads();
  if (threadIdx.x < 16) {
    const int r = threadIdx.x;
    const float m = stat[r][0] / (float)HID;
    const float var = fmaxf((stat[r][1] - (float)HID * m * m) / (float)(HID - 1), 0.f);
    mrow[r] = m; irow[r] = 1.0f / (sqrtf(var) + LN_EPS);
  }
  __syncthreads();
#pragma unroll
  for (int nt = 0; nt < 4; ++nt) {
    const int col = n0 + nt * 16 + col_l;
    const float gg = lng[col], gb = lnb[col];
#pragma unroll
    for (int i = 0; i < 8; ++i) {
      const int rl  = half * 8 + i;
      const int row = m0 + rl;
      const float vn = gg * (acc[nt][i] - mrow[rl]) * irow[rl] + gb;
      const float i3 = gi[((size_t)t * BATCH + row) * G3 + G2 + col];
      const float th = tanhf(i3 + vn);
      const float z  = zbuf[row * HID + col];
      const float ho = hf[row * HID + col];
      const float hn = (1.0f - z) * ho + z * th;
      hf[row * HID + col] = hn;
      hb[row * HID + col] = (bf16_t)hn;
      out[((size_t)t * BATCH + row) * HID + col] = hn;
      if (t == T_STEPS - 1)
        out[(size_t)T_STEPS * BATCH * HID + row * HID + col] = hn;  // h_last
    }
  }
}

// ---------------------------------------------------------------------------
extern "C" void kernel_launch(void* const* d_in, const int* in_sizes, int n_in,
                              void* d_out, int out_size, void* d_ws, size_t ws_size,
                              hipStream_t stream) {
  (void)in_sizes; (void)n_in; (void)out_size; (void)ws_size;
  const float* x       = (const float*)d_in[0];
  const float* w_ih    = (const float*)d_in[1];
  const float* w_hh    = (const float*)d_in[2];
  const float* b_ih    = (const float*)d_in[3];
  const float* b_hh    = (const float*)d_in[4];
  const float* ln_ih_g = (const float*)d_in[5];
  const float* ln_ih_b = (const float*)d_in[6];
  const float* ln_hh_g = (const float*)d_in[7];
  const float* ln_hh_b = (const float*)d_in[8];
  const float* ln_ih3_g = (const float*)d_in[9];
  const float* ln_ih3_b = (const float*)d_in[10];
  const float* ln_hh3_g = (const float*)d_in[11];
  const float* ln_hh3_b = (const float*)d_in[12];
  float* out = (float*)d_out;

  // workspace layout (bytes)
  char* ws = (char*)d_ws;
  float*  gi   = (float*)(ws);                      // ROWS*G3*4      = 402,653,184
  bf16_t* xb   = (bf16_t*)(ws + 402653184ULL);      // ROWS*HID*2     =  67,108,864
  bf16_t* wihb = (bf16_t*)(ws + 469762048ULL);      // G3*HID*2       =   6,291,456
  bf16_t* whhb = (bf16_t*)(ws + 476053504ULL);      // G3*HID*2       =   6,291,456
  float*  hf   = (float*)(ws + 482344960ULL);       // BATCH*HID*4    =     262,144
  float*  zb   = (float*)(ws + 482607104ULL);       // BATCH*HID*4    =     262,144
  bf16_t* hb   = (bf16_t*)(ws + 482869248ULL);      // BATCH*HID*2    =     131,072
  bf16_t* rhb  = (bf16_t*)(ws + 483000320ULL);      // BATCH*HID*2    =     131,072

  // 1) bf16 conversions
  {
    const int nx4 = ROWS * HID / 4;                 // 8,388,608
    cvt_f32_bf16_kernel<<<nx4 / 256, 256, 0, stream>>>(x, xb, nx4);
    const int nw4 = G3 * HID / 4;                   // 786,432
    cvt_f32_bf16_kernel<<<nw4 / 256, 256, 0, stream>>>(w_ih, wihb, nw4);
    cvt_f32_bf16_kernel<<<nw4 / 256, 256, 0, stream>>>(w_hh, whhb, nw4);
  }
  // 2) h0 = 0
  init_h_kernel<<<(BATCH * HID) / 256, 256, 0, stream>>>(hf, hb, BATCH * HID);

  // 3) input projection GEMM: (2048 m-tiles) x (48 n-chunks) waves, 8 waves/blk
  gemm_gi_kernel<<<(ROWS / 16) * (G3 / 64) / 8, 256, 0, stream>>>(xb, wihb, gi);

  // 4) bias + LayerNorm of gi (in place)
  ln_gi_kernel<<<ROWS, 256, 0, stream>>>(gi, b_ih, ln_ih_g, ln_ih_b,
                                         ln_ih3_g, ln_ih3_b);

  // 5) sequential scan: 2 fused kernels per step
  for (int t = 0; t < T_STEPS; ++t) {
    step_zr_kernel<<<BATCH / 16, 512, 0, stream>>>(hb, whhb, gi, b_hh,
                                                   ln_hh_g, ln_hh_b, hf, zb, rhb, t);
    step_h_kernel<<<BATCH / 16, 512, 0, stream>>>(rhb, whhb + (size_t)G2 * HID, gi,
                                                  b_hh + G2, ln_hh3_g, ln_hh3_b,
                                                  zb, hf, hb, out, t);
  }
}
